// ImplicitTransformer_41979010351711
// MI455X (gfx1250) — matrix operand order repaired and checked by reading, not compiled, not run on hardware
//
#include <hip/hip_runtime.h>
#include <hip/hip_bf16.h>
#include <stdint.h>

#define BATCH   8
#define SEQ     256
#define TP      257
#define ROWS    (BATCH * TP)      // 2056
#define ROWS_PAD 2176             // multiple of 128 (workgroup M-tile)
#define KDIM    2048              // [z(1792) | u(256)]
#define NPRE    1536
#define RELU_N  1024
#define LN_N    512
#define LN_GRP  256
#define NQKV    768
#define HEADS   8
#define HDIM    32
#define EMB     256
#define CLS_IN  1792
#define NCLS    16
#define N_ITERS 12
#define EPS     1e-5f

typedef __attribute__((ext_vector_type(16))) __bf16 v16bf;
typedef __attribute__((ext_vector_type(8)))  __bf16 v8bf;
typedef __attribute__((ext_vector_type(8)))  float  v8f;
typedef __attribute__((ext_vector_type(4))) unsigned int u32x4;
typedef __attribute__((ext_vector_type(8))) int i32x8;
typedef __attribute__((ext_vector_type(4))) int i32x4;

// ---------------------------------------------------------------- utilities
__global__ void k_zero32(uint32_t* p, long n) {
    long i = (long)blockIdx.x * blockDim.x + threadIdx.x;
    long stride = (long)gridDim.x * blockDim.x;
    for (; i < n; i += stride) p[i] = 0u;
}

__global__ void k_f32_to_bf16(const float* __restrict__ src,
                              __hip_bfloat16* __restrict__ dst, long n) {
    long i = (long)blockIdx.x * blockDim.x + threadIdx.x;
    if (i < n) dst[i] = __float2bfloat16(src[i]);
}

// ------------------------------------------------- embedding + pos-emb -> u
__global__ void k_embed(const int* __restrict__ x, const float* __restrict__ emb,
                        const float* __restrict__ cls_tok,
                        __hip_bfloat16* __restrict__ Xcat) {
    int idx = blockIdx.x * blockDim.x + threadIdx.x;
    if (idx >= ROWS * EMB) return;
    int row = idx / EMB, e = idx % EMB;
    int b = row / TP, tp = row % TP;
    float v;
    if (tp == 0) {
        v = cls_tok[e];
    } else {
        int tok = x[b * SEQ + (tp - 1)];
        float pos = (float)(tp - 1);
        int i2 = e >> 1;
        float ang = pos * __expf(-(float)i2 * (9.210340371976184f / 128.0f));
        float pe = (e & 1) ? __cosf(ang) : __sinf(ang);
        v = emb[(long)tok * EMB + e] + pe;
    }
    Xcat[(long)row * KDIM + CLS_IN + e] = __float2bfloat16(v);
}

// ------------------------------------------ TDM: 2D tile (rows x 32 bf16) -> LDS
// D# per CDNA5 ISA ch.8: group0 = {ctrl, lds_addr, global_addr, type=2},
// group1 = {data_size, tensor dims, tile dims, dim0 stride}.
__device__ __forceinline__ void tdm_load_2d(uint32_t lds_off, const void* gaddr,
                                            int rows, int row_stride_elems) {
    uint64_t ga = (uint64_t)(uintptr_t)gaddr;
    u32x4 g0;
    g0[0] = 1u;                                        // count=1, user mode
    g0[1] = lds_off;                                   // lds_addr (bits 63:32)
    g0[2] = (uint32_t)ga;                              // global_addr lo
    g0[3] = (uint32_t)((ga >> 32) & 0x1FFFFFFu)        // global_addr hi (56:32)
            | (2u << 30);                              // type = 2 (image)
    i32x8 g1;
    g1[0] = (int)(1u << 16);                           // data_size = 1 (2 bytes)
    g1[1] = (int)(32u << 16);                          // tensor_dim0 = 32 (lo16)
    g1[2] = (int)((uint32_t)rows << 16);               // td0 hi=0 | tensor_dim1 lo16
    g1[3] = (int)(32u << 16);                          // td1 hi=0 | tile_dim0 = 32
    g1[4] = (int)(uint32_t)rows;                       // tile_dim1 = rows, tile_dim2 = 0
    g1[5] = (int)(uint32_t)row_stride_elems;           // tensor_dim0_stride lo32
    g1[6] = 0;                                         // stride0 hi | stride1 lo (2D: unused)
    g1[7] = 0;
    i32x4 z4 = {0, 0, 0, 0};
#if defined(__clang_major__) && (__clang_major__ >= 23)
    i32x8 z8 = {0, 0, 0, 0, 0, 0, 0, 0};
    __builtin_amdgcn_tensor_load_to_lds(g0, g1, z4, z4, z8, 0);
#else
    __builtin_amdgcn_tensor_load_to_lds(g0, g1, z4, z4, 0);
#endif
}

// ------------------------------------------------------------ WMMA GEMM (TDM)
// out[m,n] = sum_k X[m,k]*W[n,k] + bias[n].
// Workgroup: 8 waves -> 128M x 64N tile; wave (4x2 grid) -> 32x32 via 2x2 WMMA
// tiles. K staged 32 at a time into double-buffered LDS by the Tensor Data
// Mover (wave 0 issues descriptors), overlapped with WMMA compute.
__global__ __launch_bounds__(256) void k_gemm_bf16_tdm(
    const __hip_bfloat16* __restrict__ Xh, int ldx,
    const __hip_bfloat16* __restrict__ Wh, int K,
    const float* __restrict__ bias,
    float* __restrict__ out, int ldo,
    int nbn, int valid_rows) {
    const int lane = threadIdx.x & 31;
    const int wave = threadIdx.x >> 5;
    const int lm = lane & 15, hi = lane >> 4;
    const int mb = (wave & 3) * 32;    // wave M offset in tile
    const int nb = (wave >> 2) * 32;   // wave N offset in tile
    const int bn = blockIdx.x % nbn;
    const int bm = blockIdx.x / nbn;
    const int row0 = bm * 128, n0 = bn * 64;

    __shared__ __hip_bfloat16 Xs[2][128 * 32];   // 16 KB
    __shared__ __hip_bfloat16 Ws[2][64 * 32];    //  8 KB

    const char* xtile = (const char*)(Xh) + (size_t)row0 * ldx * 2;
    const char* wtile = (const char*)(Wh) + (size_t)n0 * K * 2;

    v8f acc00 = {}, acc01 = {}, acc10 = {}, acc11 = {};

    const int nk = K >> 5;
    if (wave == 0) {   // wave-uniform branch: EXEC stays all-ones
        tdm_load_2d((uint32_t)(uintptr_t)&Xs[0][0], xtile, 128, ldx);
        tdm_load_2d((uint32_t)(uintptr_t)&Ws[0][0], wtile, 64, K);
    }
    __builtin_amdgcn_s_wait_tensorcnt(0);
    __syncthreads();

    for (int kk = 0; kk < nk; ++kk) {
        const int cur = kk & 1;
        if (kk + 1 < nk && wave == 0) {
            const size_t koff = (size_t)(kk + 1) * 64;  // 32 bf16 = 64 bytes
            tdm_load_2d((uint32_t)(uintptr_t)&Xs[cur ^ 1][0], xtile + koff, 128, ldx);
            tdm_load_2d((uint32_t)(uintptr_t)&Ws[cur ^ 1][0], wtile + koff, 64, K);
        }

        const __bf16* xs  = reinterpret_cast<const __bf16*>(&Xs[cur][0]);
        const __bf16* wsm = reinterpret_cast<const __bf16*>(&Ws[cur][0]);
        // A fragments: lane=row, 16 contiguous K per lane half (k = 16*hi + s)
        v8bf a0l = *reinterpret_cast<const v8bf*>(xs + (mb + lm) * 32 + hi * 16);
        v8bf a0h = *reinterpret_cast<const v8bf*>(xs + (mb + lm) * 32 + 8 + hi * 16);
        v8bf a1l = *reinterpret_cast<const v8bf*>(xs + (mb + 16 + lm) * 32 + hi * 16);
        v8bf a1h = *reinterpret_cast<const v8bf*>(xs + (mb + 16 + lm) * 32 + 8 + hi * 16);
        v16bf a0 = __builtin_shufflevector(a0l, a0h, 0,1,2,3,4,5,6,7,8,9,10,11,12,13,14,15);
        v16bf a1 = __builtin_shufflevector(a1l, a1h, 0,1,2,3,4,5,6,7,8,9,10,11,12,13,14,15);
        // B fragments (32x16 = W^T tile): lane=col, 16 contiguous K
        v16bf b0 = *reinterpret_cast<const v16bf*>(wsm + (nb + lm) * 32 + hi * 16);
        v16bf b1 = *reinterpret_cast<const v16bf*>(wsm + (nb + 16 + lm) * 32 + hi * 16);

        acc00 = __builtin_amdgcn_wmma_f32_16x16x32_bf16(false, a0, false, b0, (short)0, acc00, false, false);
        acc01 = __builtin_amdgcn_wmma_f32_16x16x32_bf16(false, a0, false, b1, (short)0, acc01, false, false);
        acc10 = __builtin_amdgcn_wmma_f32_16x16x32_bf16(false, a1, false, b0, (short)0, acc10, false, false);
        acc11 = __builtin_amdgcn_wmma_f32_16x16x32_bf16(false, a1, false, b1, (short)0, acc11, false, false);

        if (kk + 1 < nk) {
            __builtin_amdgcn_s_wait_tensorcnt(0);
            __syncthreads();
        }
    }

    // C/D layout: lane 0-15 -> N=lane, M=vgpr; lane 16-31 -> M=8+vgpr
    const int nA = n0 + nb + lm;
    const int nB = n0 + nb + 16 + lm;
    const float bvA = bias ? bias[nA] : 0.0f;
    const float bvB = bias ? bias[nB] : 0.0f;
#pragma unroll
    for (int i = 0; i < 8; ++i) {
        int mA = row0 + mb + hi * 8 + i;
        int mB = row0 + mb + 16 + hi * 8 + i;
        if (mA < valid_rows) {
            out[(size_t)mA * ldo + nA] = acc00[i] + bvA;
            out[(size_t)mA * ldo + nB] = acc01[i] + bvB;
        }
        if (mB < valid_rows) {
            out[(size_t)mB * ldo + nA] = acc10[i] + bvA;
            out[(size_t)mB * ldo + nB] = acc11[i] + bvB;
        }
    }
}

// ----------------------------------------------------------- ReLU epilogue
__global__ void k_relu(const float* __restrict__ pre,
                       __hip_bfloat16* __restrict__ Xcat) {
    long idx = (long)blockIdx.x * blockDim.x + threadIdx.x;
    if (idx >= (long)ROWS * RELU_N) return;
    long row = idx / RELU_N, c = idx % RELU_N;
    float v = pre[row * NPRE + c];
    Xcat[row * KDIM + c] = __float2bfloat16(fmaxf(v, 0.0f));
}

// -------------------------------------------------- LayerNorm over 256 elems
__global__ __launch_bounds__(256) void k_layernorm(
    const float* __restrict__ pre, __hip_bfloat16* __restrict__ Xcat) {
    int blk = blockIdx.x;            // token*2 + group
    int t = blk >> 1, g = blk & 1;
    int tid = threadIdx.x;
    float v = pre[(long)t * NPRE + RELU_N + g * LN_GRP + tid];
    __shared__ float s1[256], s2[256];
    s1[tid] = v; s2[tid] = v * v;
    __syncthreads();
    for (int s = 128; s > 0; s >>= 1) {
        if (tid < s) { s1[tid] += s1[tid + s]; s2[tid] += s2[tid + s]; }
        __syncthreads();
    }
    float mean = s1[0] * (1.0f / LN_GRP);
    float var  = s2[0] * (1.0f / LN_GRP) - mean * mean;
    float rs = rsqrtf(var + EPS);
    Xcat[(long)t * KDIM + RELU_N + g * LN_GRP + tid] =
        __float2bfloat16((v - mean) * rs);
}

// ------------------------------------------- attention (one block per (b,h))
__global__ __launch_bounds__(256) void k_attn(
    const float* __restrict__ qkv, const int* __restrict__ seq_lens,
    __hip_bfloat16* __restrict__ Xcat) {
    int b = blockIdx.x / HEADS, h = blockIdx.x % HEADS;
    __shared__ __hip_bfloat16 ksm[TP * HDIM];
    __shared__ __hip_bfloat16 vsm[TP * HDIM];
    for (int idx = threadIdx.x; idx < TP * HDIM; idx += 256) {
        int t = idx / HDIM, d = idx % HDIM;
        long base = (long)(b * TP + t) * NQKV + h * (3 * HDIM);
        ksm[idx] = __float2bfloat16(qkv[base + HDIM + d]);
        vsm[idx] = __float2bfloat16(qkv[base + 2 * HDIM + d]);
    }
    __syncthreads();
    int slen = seq_lens[b] + 1;      // keys j >= slen masked to -1e9 -> weight 0
    if (slen > TP) slen = TP;
    const float scale = 0.17677669529663688f;  // 1/sqrt(32)
    for (int tq = threadIdx.x; tq < TP; tq += 256) {
        long qbase = (long)(b * TP + tq) * NQKV + h * (3 * HDIM);
        float q[HDIM];
#pragma unroll
        for (int d = 0; d < HDIM; ++d) q[d] = qkv[qbase + d] * scale;
        float m = -3.0e38f, l = 0.0f;
        float acc[HDIM];
#pragma unroll
        for (int d = 0; d < HDIM; ++d) acc[d] = 0.0f;
        for (int j = 0; j < slen; ++j) {
            float s = 0.0f;
#pragma unroll
            for (int d = 0; d < HDIM; ++d)
                s += q[d] * __bfloat162float(ksm[j * HDIM + d]);
            if (s <= m) {
                float p = __expf(s - m);
                l += p;
#pragma unroll
                for (int d = 0; d < HDIM; ++d)
                    acc[d] += p * __bfloat162float(vsm[j * HDIM + d]);
            } else {
                float c = __expf(m - s);
                l = l * c + 1.0f;
#pragma unroll
                for (int d = 0; d < HDIM; ++d)
                    acc[d] = acc[d] * c + __bfloat162float(vsm[j * HDIM + d]);
                m = s;
            }
        }
        float inv = 1.0f / l;
        long orow = (long)(b * TP + tq) * KDIM + RELU_N + LN_N + h * HDIM;
#pragma unroll
        for (int d = 0; d < HDIM; ++d)
            Xcat[orow + d] = __float2bfloat16(acc[d] * inv);
    }
}

// ----------------------------------------------------- final classifier 8x16
__global__ void k_cls(const __hip_bfloat16* __restrict__ Xcat,
                      const float* __restrict__ Wcls,
                      const float* __restrict__ bcls,
                      float* __restrict__ out) {
    int t = threadIdx.x;
    if (t >= BATCH * NCLS) return;
    int b = t / NCLS, c = t % NCLS;
    const __hip_bfloat16* zrow = Xcat + (long)(b * TP) * KDIM;
    float s = bcls[c];
    for (int j = 0; j < CLS_IN; ++j)
        s += __bfloat162float(zrow[j]) * Wcls[(long)c * CLS_IN + j];
    out[t] = s;
}

// ---------------------------------------------------------------- launcher
extern "C" void kernel_launch(void* const* d_in, const int* in_sizes, int n_in,
                              void* d_out, int out_size, void* d_ws, size_t ws_size,
                              hipStream_t stream) {
    const int*   x        = (const int*)d_in[0];
    const int*   seq_lens = (const int*)d_in[1];
    const float* emb      = (const float*)d_in[2];
    const float* cls_tok  = (const float*)d_in[3];
    const float* A        = (const float*)d_in[4];
    const float* A_qkv    = (const float*)d_in[5];
    const float* bvec     = (const float*)d_in[6];
    const float* Wcls     = (const float*)d_in[7];
    const float* bcls     = (const float*)d_in[8];
    float* out = (float*)d_out;

    char* ws = (char*)d_ws;
    size_t off = 0;
    auto take = [&](size_t bytes) {
        char* p = ws + off;
        off = (off + bytes + 255) & ~(size_t)255;
        return p;
    };
    __hip_bfloat16* Xcat = (__hip_bfloat16*)take((size_t)ROWS_PAD * KDIM * 2);
    __hip_bfloat16* Abf  = (__hip_bfloat16*)take((size_t)NPRE * KDIM * 2);
    __hip_bfloat16* Aqkv = (__hip_bfloat16*)take((size_t)NQKV * LN_N * 2);
    float* pre  = (float*)take((size_t)ROWS * NPRE * 4);
    float* qkvb = (float*)take((size_t)ROWS * NQKV * 4);

    // zero Xcat: z0 = 0 and padding rows 2056..2175 stay zero forever
    long nz = (long)ROWS_PAD * KDIM / 2;  // as u32
    k_zero32<<<1024, 256, 0, stream>>>((uint32_t*)Xcat, nz);

    long nA = (long)NPRE * KDIM;
    k_f32_to_bf16<<<(int)((nA + 255) / 256), 256, 0, stream>>>(A, Abf, nA);
    long nQ = (long)NQKV * LN_N;
    k_f32_to_bf16<<<(int)((nQ + 255) / 256), 256, 0, stream>>>(A_qkv, Aqkv, nQ);

    int ne = ROWS * EMB;
    k_embed<<<(ne + 255) / 256, 256, 0, stream>>>(x, emb, cls_tok, Xcat);

    const int nbm  = ROWS_PAD / 128;  // 17
    const int nbn1 = NPRE / 64;       // 24
    const int nbn2 = NQKV / 64;       // 12
    long nrelu = (long)ROWS * RELU_N;

    for (int it = 0; it < N_ITERS; ++it) {
        k_gemm_bf16_tdm<<<nbm * nbn1, 256, 0, stream>>>(
            Xcat, KDIM, Abf, KDIM, bvec, pre, NPRE, nbn1, ROWS);
        k_relu<<<(int)((nrelu + 255) / 256), 256, 0, stream>>>(pre, Xcat);
        k_layernorm<<<ROWS * 2, 256, 0, stream>>>(pre, Xcat);
        k_gemm_bf16_tdm<<<nbm * nbn2, 256, 0, stream>>>(
            Xcat + RELU_N, KDIM, Aqkv, LN_N, nullptr, qkvb, NQKV, nbn2, ROWS);
        k_attn<<<BATCH * HEADS, 256, 0, stream>>>(qkvb, seq_lens, Xcat);
    }
    k_cls<<<1, 128, 0, stream>>>(Xcat, Wcls, bcls, out);
}
